// LGEB_adapt_32890859553317
// MI455X (gfx1250) — matrix operand order repaired and checked by reading, not compiled, not run on hardware
//
#include <hip/hip_runtime.h>
#include <hip/hip_bf16.h>

// ---------------------------------------------------------------------------
// CDNA5 (gfx1250, wave32) WMMA helpers
// ---------------------------------------------------------------------------
typedef __attribute__((ext_vector_type(16))) __bf16 v16bf;
typedef __attribute__((ext_vector_type(8)))  float  v8f;

union BfVec { v16bf v; unsigned int u[8]; unsigned short h[16]; };

__device__ __forceinline__ unsigned short f2bf(float x) {
  unsigned int u = __float_as_uint(x);
  unsigned int r = (u + 0x7FFFu + ((u >> 16) & 1u)) >> 16; // RNE
  return (unsigned short)r;
}
__device__ __forceinline__ float bf2f(unsigned short h) {
  return __uint_as_float(((unsigned int)h) << 16);
}

__device__ __forceinline__ v8f wmma_bf16(v16bf a, v16bf b, v8f c) {
  return __builtin_amdgcn_wmma_f32_16x16x32_bf16(false, a, false, b,
                                                 (short)0, c, false, false);
}

// A-tile (16x32 bf16) from row-major LDS ushort array.
// Layout (contiguous, matches packed B): lanes 0-15 row M=lane, dwords 0..7 =
// K0..15; lanes 16-31 same rows, K16..31.
__device__ __forceinline__ v16bf ldsA(const unsigned short* base, int row0,
                                      int strideH, int kBase) {
  int lane = threadIdx.x & 31;
  int m = lane & 15;
  int k0 = (lane >> 4) * 8; // dword offset within K-group
  const unsigned int* p =
      (const unsigned int*)(base + (row0 + m) * strideH + kBase);
  BfVec a;
#pragma unroll
  for (int d = 0; d < 4; ++d) { a.u[d] = p[k0 + d]; a.u[4 + d] = p[k0 + 4 + d]; }
  return a.v;
}

// B-tile from pre-packed LDS: [tile][lane][8 dwords] -> 2x ds_load_b128/lane.
__device__ __forceinline__ v16bf ldsB(const unsigned int* bp, int tileIdx) {
  int lane = threadIdx.x & 31;
  const unsigned int* p = bp + (tileIdx * 32 + lane) * 8;
  BfVec b;
#pragma unroll
  for (int d = 0; d < 8; ++d) b.u[d] = p[d];
  return b.v;
}

// Pack global f32 weight W (Kreal x Nreal, leading dim ldw) into this file's contiguous-k wave32
// B-operand layout (32x16 bf16 tiles): lane<16 holds K=kt*32+e, lane>=16 holds
// K=kt*32+16+e, N = nt*16 + lane%16. Zero-padded.
__device__ __forceinline__ void packB(unsigned short* dst, const float* W,
                                      int Kreal, int Nreal, int ldw, int ntK,
                                      int ntN, int nthreads, int tid) {
  int total = ntN * ntK * 512;
  for (int idx = tid; idx < total; idx += nthreads) {
    int e = idx & 15;
    int lane = (idx >> 4) & 31;
    int tile = idx >> 9;
    int kt = tile % ntK;
    int nt = tile / ntK;
    int k = kt * 32 + ((lane >> 4) << 4) + e;
    int n = nt * 16 + (lane & 15);
    float w = (k < Kreal && n < Nreal) ? W[k * ldw + n] : 0.0f;
    dst[idx] = f2bf(w);
  }
}

__device__ __forceinline__ float psi_f(float x) {
  return copysignf(log1pf(fabsf(x)), x);
}

// ---------------------------------------------------------------------------
// Kernel A: ABrow[r][0:72] = h[r]@We1[0:72,:], ABrow[r][72:144] = h[r]@We1[72:144,:]
// one GEMM 4096 x 96(K pad) x 144, blocks of 64 rows, 4 waves.
// ---------------------------------------------------------------------------
__global__ __launch_bounds__(128) void kA(const float* __restrict__ h,
                                          const float* __restrict__ We1,
                                          float* __restrict__ ABrow) {
  __shared__ __align__(16) unsigned short hs[64 * 96];
  __shared__ __align__(16) unsigned short bp[9 * 3 * 512];
  int tid = threadIdx.x;
  int rowBase = blockIdx.x * 64;
  for (int idx = tid; idx < 64 * 96; idx += 128) {
    int r = idx / 96, c = idx % 96;
    hs[idx] = f2bf((c < 72) ? h[(rowBase + r) * 72 + c] : 0.0f);
  }
  for (int idx = tid; idx < 9 * 3 * 512; idx += 128) {
    int e = idx & 15, lane = (idx >> 4) & 31;
    int tile = idx >> 9, kt = tile % 3, nt = tile / 3;
    int k = kt * 32 + ((lane >> 4) << 4) + e;
    int n = nt * 16 + (lane & 15);
    float w = 0.0f;
    if (k < 72) w = (n < 72) ? We1[k * 72 + n] : We1[(72 + k) * 72 + (n - 72)];
    bp[idx] = f2bf(w);
  }
  __syncthreads();
  int wave = tid >> 5, lane = tid & 31;
  v8f zero = {0.f, 0.f, 0.f, 0.f, 0.f, 0.f, 0.f, 0.f};
  v8f acc[9];
#pragma unroll
  for (int nt = 0; nt < 9; ++nt) acc[nt] = zero;
  for (int kt = 0; kt < 3; ++kt) {
    v16bf a = ldsA(hs, wave * 16, 96, kt * 32);
#pragma unroll
    for (int nt = 0; nt < 9; ++nt)
      acc[nt] = wmma_bf16(a, ldsB((const unsigned int*)bp, nt * 3 + kt), acc[nt]);
  }
  int colh = lane & 15, rh = (lane >> 4) * 8;
#pragma unroll
  for (int nt = 0; nt < 9; ++nt)
#pragma unroll
    for (int v = 0; v < 8; ++v) {
      int row = rowBase + wave * 16 + v + rh;
      ABrow[row * 144 + nt * 16 + colh] = acc[nt][v];
    }
}

// ---------------------------------------------------------------------------
// Kernel P: per-pair Minkowski norms/dots -> nd[pair][2]; zero BN1 stats.
// grid (i=256, b=16), 256 threads = j.
// ---------------------------------------------------------------------------
__global__ __launch_bounds__(256) void kP(const float* __restrict__ p,
                                          float* __restrict__ nd,
                                          float* __restrict__ stats) {
  int i = blockIdx.x, b = blockIdx.y, j = threadIdx.x;
  if (blockIdx.x == 0 && blockIdx.y == 0 && j < 144) stats[j] = 0.0f;
  const float* pi = p + (b * 256 + i) * 4;
  const float* pj = p + (b * 256 + j) * 4;
  float d0 = pi[0] - pj[0], d1 = pi[1] - pj[1], d2 = pi[2] - pj[2], d3 = pi[3] - pj[3];
  float ssum = d0 * d0 + d1 * d1 + d2 * d2 + d3 * d3;
  float n = psi_f(ssum - 2.0f * d0 * d0);
  float pr0 = pi[0] * pj[0], pr1 = pi[1] * pj[1], pr2 = pi[2] * pj[2], pr3 = pi[3] * pj[3];
  float d = psi_f(2.0f * pr0 - (pr0 + pr1 + pr2 + pr3));
  long idx = ((long)(b * 256 + i) * 256 + j) * 2;
  nd[idx] = n;
  nd[idx + 1] = d;
}

// ---------------------------------------------------------------------------
// Kernel S: BN1 statistics. grid (c=72, chunk=64); each block reduces 16384
// pairs for one channel. x = A_i[c] + B_j[c] + n*w144[c] + d*w145[c].
// ---------------------------------------------------------------------------
__global__ __launch_bounds__(256) void kS(const float* __restrict__ ABrow,
                                          const float* __restrict__ nd,
                                          const float* __restrict__ We1,
                                          float* __restrict__ stats) {
  int c = blockIdx.x, chunk = blockIdx.y, tid = threadIdx.x;
  float w1 = We1[144 * 72 + c], w2 = We1[145 * 72 + c];
  float s = 0.0f, sq = 0.0f;
  for (int k = 0; k < 64; ++k) {
    int pidx = chunk * 16384 + k * 256 + tid;
    int b = pidx >> 16, i = (pidx >> 8) & 255, j = pidx & 255;
    float A = ABrow[(b * 256 + i) * 144 + c];
    float B = ABrow[(b * 256 + j) * 144 + 72 + c];
    float n = nd[(long)pidx * 2], d = nd[(long)pidx * 2 + 1];
    float x = A + B + n * w1 + d * w2;
    s += x;
    sq += x * x;
  }
  __shared__ float rs[256], rq[256];
  rs[tid] = s; rq[tid] = sq;
  __syncthreads();
  for (int st = 128; st > 0; st >>= 1) {
    if (tid < st) { rs[tid] += rs[tid + st]; rq[tid] += rq[tid + st]; }
    __syncthreads();
  }
  if (tid == 0) { atomicAdd(&stats[c], rs[0]); atomicAdd(&stats[72 + c], rq[0]); }
}

// Kernel F1: fold BN1 into affine a1,c1.
__global__ void kF1(const float* __restrict__ stats, const float* __restrict__ g1,
                    const float* __restrict__ b1, float* __restrict__ a1c1) {
  int c = threadIdx.x;
  if (c < 72) {
    float cnt = 16.0f * 256.0f * 256.0f;
    float mu = stats[c] / cnt;
    float var = stats[72 + c] / cnt - mu * mu;
    float a = rsqrtf(var + 1e-5f) * g1[c];
    a1c1[c] = a;
    a1c1[72 + c] = b1[c] - mu * a;
  }
}

// ---------------------------------------------------------------------------
// Kernel D: fused pair pipeline. One block per (b, i-tile of 16). Loops 16
// j-tiles; 256 rows = (ii,jj). Per iter: build xhat (bf16, LDS) -> WMMA GEMM1
// (We2) -> shfl-reduced sigmoid gate -> wm back to LDS -> WMMA GEMM2 (Wx1) ->
// shfl-reduced xw -> clipped p_diff aggregation. magg/p_out written once.
// ---------------------------------------------------------------------------
__global__ __launch_bounds__(256, 1) void kD(
    const float* __restrict__ p, const float* __restrict__ ABrow,
    const float* __restrict__ nd, const float* __restrict__ We1,
    const float* __restrict__ We2, const float* __restrict__ be2,
    const float* __restrict__ Wm, const float* __restrict__ bm,
    const float* __restrict__ Wx1, const float* __restrict__ bx1,
    const float* __restrict__ Wx2, const float* __restrict__ a1c1,
    float* __restrict__ magg, float* __restrict__ pout) {
  __shared__ __align__(16) unsigned short xs[256 * 96];    // activations (bf16)
  __shared__ __align__(16) unsigned short bpE[5 * 3 * 512]; // We2 packed
  __shared__ __align__(16) unsigned short bpX[5 * 3 * 512]; // Wx1 packed
  __shared__ float Ai[16 * 72], Bj[16 * 72];
  __shared__ float piL[64], pjL[64];
  __shared__ float maggL[16 * 72];
  __shared__ float aggL[64];
  __shared__ float prm_be2[80], prm_bx1[80], prm_Wm[80], prm_Wx2[80];
  __shared__ float prm_a1[72], prm_c1[72], prm_w1[72], prm_w2[72];

  int tid = threadIdx.x;
  int it = blockIdx.x, b = blockIdx.y;
  int i0 = it * 16;
  int lane = tid & 31, wave = tid >> 5;
  int colh = lane & 15, rh = lane >> 4;

  for (int idx = tid; idx < 80; idx += 256) {
    prm_be2[idx] = (idx < 72) ? be2[idx] : 0.f;
    prm_bx1[idx] = (idx < 72) ? bx1[idx] : 0.f;
    prm_Wm[idx]  = (idx < 72) ? Wm[idx]  : 0.f;
    prm_Wx2[idx] = (idx < 72) ? Wx2[idx] : 0.f;
  }
  for (int idx = tid; idx < 72; idx += 256) {
    prm_a1[idx] = a1c1[idx];
    prm_c1[idx] = a1c1[72 + idx];
    prm_w1[idx] = We1[144 * 72 + idx];
    prm_w2[idx] = We1[145 * 72 + idx];
  }
  for (int idx = tid; idx < 16 * 72; idx += 256) {
    int ii = idx / 72, c = idx % 72;
    Ai[idx] = ABrow[(b * 256 + i0 + ii) * 144 + c];
    maggL[idx] = 0.f;
  }
  for (int idx = tid; idx < 64; idx += 256) {
    piL[idx] = p[(b * 256 + i0 + (idx >> 2)) * 4 + (idx & 3)];
    aggL[idx] = 0.f;
  }
  { // zero K-pad columns 72..95 once (stay zero thereafter)
    unsigned int* xs32 = (unsigned int*)xs;
    for (int idx = tid; idx < 256 * 12; idx += 256)
      xs32[(idx / 12) * 48 + 36 + (idx % 12)] = 0u;
  }
  packB(bpE, We2, 72, 72, 72, 3, 5, 256, tid);
  packB(bpX, Wx1, 72, 72, 72, 3, 5, 256, tid);

  float bmv = bm[0];
  v8f zero = {0.f, 0.f, 0.f, 0.f, 0.f, 0.f, 0.f, 0.f};

  for (int jt = 0; jt < 16; ++jt) {
    __syncthreads();
    int j0 = jt * 16;
    for (int idx = tid; idx < 16 * 72; idx += 256) {
      int jj = idx / 72, c = idx % 72;
      Bj[idx] = ABrow[(b * 256 + j0 + jj) * 144 + 72 + c];
    }
    for (int idx = tid; idx < 64; idx += 256)
      pjL[idx] = p[(b * 256 + j0 + (idx >> 2)) * 4 + (idx & 3)];
    if (jt < 15) // prefetch next tile's Brow rows into cache
      __builtin_prefetch(&ABrow[(b * 256 + j0 + 16) * 144 + 72], 0, 1);
    __syncthreads();

    { // phase 0b: thread tid builds row tid of xhat (bf16) — wave-local rows
      int ii = tid >> 4, jj = tid & 15;
      long pidx = ((long)(b * 256 + i0 + ii)) * 256 + (j0 + jj);
      float n = nd[pidx * 2], d = nd[pidx * 2 + 1];
      unsigned short* xr = xs + tid * 96;
      for (int c = 0; c < 72; ++c) {
        float x = Ai[ii * 72 + c] + Bj[jj * 72 + c] + n * prm_w1[c] + d * prm_w2[c];
        xr[c] = f2bf(fmaxf(prm_a1[c] * x + prm_c1[c], 0.f));
      }
    }

    // GEMM1: m = relu(xhat @ We2 + be2); each wave owns mtiles {2w, 2w+1}
    v8f acc[2][5];
#pragma unroll
    for (int m = 0; m < 2; ++m)
#pragma unroll
      for (int nt = 0; nt < 5; ++nt) acc[m][nt] = zero;
    for (int kt = 0; kt < 3; ++kt) {
      v16bf a0 = ldsA(xs, (wave * 2 + 0) * 16, 96, kt * 32);
      v16bf a1v = ldsA(xs, (wave * 2 + 1) * 16, 96, kt * 32);
#pragma unroll
      for (int nt = 0; nt < 5; ++nt) {
        v16bf bb = ldsB((const unsigned int*)bpE, nt * 3 + kt);
        acc[0][nt] = wmma_bf16(a0, bb, acc[0][nt]);
        acc[1][nt] = wmma_bf16(a1v, bb, acc[1][nt]);
      }
    }
    // epilogue 1: gate + wm -> xs (bf16), magg accumulation (all in-wave)
#pragma unroll
    for (int m = 0; m < 2; ++m) {
      int mt = wave * 2 + m; // == ii of this tile
      float mval[5][8];
#pragma unroll
      for (int nt = 0; nt < 5; ++nt)
#pragma unroll
        for (int v = 0; v < 8; ++v)
          mval[nt][v] = fmaxf(acc[m][nt][v] + prm_be2[nt * 16 + colh], 0.f);
#pragma unroll
      for (int v = 0; v < 8; ++v) {
        float pd = 0.f;
#pragma unroll
        for (int nt = 0; nt < 5; ++nt) pd += mval[nt][v] * prm_Wm[nt * 16 + colh];
#pragma unroll
        for (int o = 8; o > 0; o >>= 1) pd += __shfl_xor(pd, o, 32);
        float wv = 1.f / (1.f + __expf(-(pd + bmv)));
        int row = mt * 16 + v + rh * 8;
#pragma unroll
        for (int nt = 0; nt < 5; ++nt) {
          float wmv = mval[nt][v] * wv;
          xs[row * 96 + nt * 16 + colh] = f2bf(wmv); // pad cols write 0
          mval[nt][v] = wmv;
        }
      }
#pragma unroll
      for (int nt = 0; nt < 5; ++nt) {
        float sum = 0.f;
#pragma unroll
        for (int v = 0; v < 8; ++v) sum += mval[nt][v];
        sum += __shfl_xor(sum, 16, 32); // fold both row-halves
        int col = nt * 16 + colh;
        if (lane < 16 && col < 72) maggL[mt * 72 + col] += sum;
      }
    }

    // GEMM2: t = relu(wm @ Wx1 + bx1); same wave-local rows, no barrier needed
#pragma unroll
    for (int m = 0; m < 2; ++m)
#pragma unroll
      for (int nt = 0; nt < 5; ++nt) acc[m][nt] = zero;
    for (int kt = 0; kt < 3; ++kt) {
      v16bf a0 = ldsA(xs, (wave * 2 + 0) * 16, 96, kt * 32);
      v16bf a1v = ldsA(xs, (wave * 2 + 1) * 16, 96, kt * 32);
#pragma unroll
      for (int nt = 0; nt < 5; ++nt) {
        v16bf bb = ldsB((const unsigned int*)bpX, nt * 3 + kt);
        acc[0][nt] = wmma_bf16(a0, bb, acc[0][nt]);
        acc[1][nt] = wmma_bf16(a1v, bb, acc[1][nt]);
      }
    }
    // epilogue 2: xw = t @ Wx2, clipped p_diff aggregation
#pragma unroll
    for (int m = 0; m < 2; ++m) {
      int mt = wave * 2 + m;
#pragma unroll
      for (int v = 0; v < 8; ++v) {
        float pd = 0.f;
#pragma unroll
        for (int nt = 0; nt < 5; ++nt) {
          float tv = fmaxf(acc[m][nt][v] + prm_bx1[nt * 16 + colh], 0.f);
          pd += tv * prm_Wx2[nt * 16 + colh];
        }
#pragma unroll
        for (int o = 8; o > 0; o >>= 1) pd += __shfl_xor(pd, o, 32);
        int jj = v + rh * 8;
        if ((lane & 15) < 4) {
          int comp = lane & 3;
          float diff = piL[mt * 4 + comp] - pjL[jj * 4 + comp];
          float contrib = fminf(fmaxf(diff * pd, -100.f), 100.f);
          atomicAdd(&aggL[mt * 4 + comp], contrib);
        }
      }
    }
  }

  __syncthreads();
  for (int idx = tid; idx < 16 * 72; idx += 256) {
    int ii = idx / 72, c = idx % 72;
    magg[(b * 256 + i0 + ii) * 72 + c] = maggL[idx];
  }
  for (int idx = tid; idx < 64; idx += 256) {
    int ii = idx >> 2, comp = idx & 3;
    long g = (long)(b * 256 + i0 + ii) * 4 + comp;
    pout[g] = p[g] + (aggL[idx] * (1.f / 256.f)) * 0.001f;
  }
}

// ---------------------------------------------------------------------------
// Kernel E1: y = [h | magg | s] @ Wh1 + bh1  (4096 x 160(K pad) x 72)
// ---------------------------------------------------------------------------
__global__ __launch_bounds__(128) void kE1(const float* __restrict__ h,
                                           const float* __restrict__ magg,
                                           const float* __restrict__ s,
                                           const float* __restrict__ Wh1,
                                           const float* __restrict__ bh1,
                                           float* __restrict__ y) {
  __shared__ __align__(16) unsigned short hs[64 * 160];
  __shared__ __align__(16) unsigned short bp[5 * 5 * 512];
  int tid = threadIdx.x;
  int rowBase = blockIdx.x * 64;
  for (int idx = tid; idx < 64 * 160; idx += 128) {
    int r = idx / 160, c = idx % 160;
    int row = rowBase + r;
    float v = 0.f;
    if (c < 72) v = h[row * 72 + c];
    else if (c < 144) v = magg[row * 72 + (c - 72)];
    else if (c < 157) v = s[row * 13 + (c - 144)];
    hs[idx] = f2bf(v);
  }
  packB(bp, Wh1, 157, 72, 72, 5, 5, 128, tid);
  __syncthreads();
  int wave = tid >> 5, lane = tid & 31;
  v8f zero = {0.f, 0.f, 0.f, 0.f, 0.f, 0.f, 0.f, 0.f};
  v8f acc[5];
#pragma unroll
  for (int nt = 0; nt < 5; ++nt) acc[nt] = zero;
  for (int kt = 0; kt < 5; ++kt) {
    v16bf a = ldsA(hs, wave * 16, 160, kt * 32);
#pragma unroll
    for (int nt = 0; nt < 5; ++nt)
      acc[nt] = wmma_bf16(a, ldsB((const unsigned int*)bp, nt * 5 + kt), acc[nt]);
  }
  int colh = lane & 15, rh = (lane >> 4) * 8;
#pragma unroll
  for (int nt = 0; nt < 5; ++nt)
#pragma unroll
    for (int v = 0; v < 8; ++v) {
      int col = nt * 16 + colh;
      if (col < 72) {
        int row = rowBase + wave * 16 + v + rh;
        y[row * 72 + col] = acc[nt][v] + bh1[col];
      }
    }
}

// Kernel E2: BN2 per-channel stats + affine fold (one block per channel).
__global__ __launch_bounds__(256) void kE2(const float* __restrict__ y,
                                           const float* __restrict__ gh,
                                           const float* __restrict__ bh,
                                           float* __restrict__ a2c2) {
  int c = blockIdx.x, tid = threadIdx.x;
  float s = 0.f, q = 0.f;
  for (int r = tid; r < 4096; r += 256) {
    float v = y[r * 72 + c];
    s += v; q += v * v;
  }
  __shared__ float rs[256], rq[256];
  rs[tid] = s; rq[tid] = q;
  __syncthreads();
  for (int st = 128; st > 0; st >>= 1) {
    if (tid < st) { rs[tid] += rs[tid + st]; rq[tid] += rq[tid + st]; }
    __syncthreads();
  }
  if (tid == 0) {
    float mu = rs[0] / 4096.f;
    float var = rq[0] / 4096.f - mu * mu;
    float a = rsqrtf(var + 1e-5f) * gh[c];
    a2c2[c] = a;
    a2c2[72 + c] = bh[c] - mu * a;
  }
}

// Kernel E4: h_out = h + relu(bn(y)) @ Wh2 + bh2
__global__ __launch_bounds__(128) void kE4(const float* __restrict__ h,
                                           const float* __restrict__ y,
                                           const float* __restrict__ a2c2,
                                           const float* __restrict__ Wh2,
                                           const float* __restrict__ bh2,
                                           float* __restrict__ hout) {
  __shared__ __align__(16) unsigned short ys[64 * 96];
  __shared__ __align__(16) unsigned short bp[5 * 3 * 512];
  __shared__ float a2s[72], c2s[72];
  int tid = threadIdx.x;
  int rowBase = blockIdx.x * 64;
  for (int idx = tid; idx < 72; idx += 128) { a2s[idx] = a2c2[idx]; c2s[idx] = a2c2[72 + idx]; }
  __syncthreads();
  for (int idx = tid; idx < 64 * 96; idx += 128) {
    int r = idx / 96, c = idx % 96;
    float v = 0.f;
    if (c < 72) v = fmaxf(a2s[c] * y[(rowBase + r) * 72 + c] + c2s[c], 0.f);
    ys[idx] = f2bf(v);
  }
  packB(bp, Wh2, 72, 72, 72, 3, 5, 128, tid);
  __syncthreads();
  int wave = tid >> 5, lane = tid & 31;
  v8f zero = {0.f, 0.f, 0.f, 0.f, 0.f, 0.f, 0.f, 0.f};
  v8f acc[5];
#pragma unroll
  for (int nt = 0; nt < 5; ++nt) acc[nt] = zero;
  for (int kt = 0; kt < 3; ++kt) {
    v16bf a = ldsA(ys, wave * 16, 96, kt * 32);
#pragma unroll
    for (int nt = 0; nt < 5; ++nt)
      acc[nt] = wmma_bf16(a, ldsB((const unsigned int*)bp, nt * 3 + kt), acc[nt]);
  }
  int colh = lane & 15, rh = (lane >> 4) * 8;
#pragma unroll
  for (int nt = 0; nt < 5; ++nt)
#pragma unroll
    for (int v = 0; v < 8; ++v) {
      int col = nt * 16 + colh;
      if (col < 72) {
        int row = rowBase + wave * 16 + v + rh;
        hout[row * 72 + col] = h[row * 72 + col] + acc[nt][v] + bh2[col];
      }
    }
}

// ---------------------------------------------------------------------------
// Host-side launch. Workspace layout (floats): ABrow | nd | stats | a1c1 |
// magg | y | a2c2  -> ~13.1 MB total.
// ---------------------------------------------------------------------------
extern "C" void kernel_launch(void* const* d_in, const int* in_sizes, int n_in,
                              void* d_out, int out_size, void* d_ws, size_t ws_size,
                              hipStream_t stream) {
  const float* h   = (const float*)d_in[0];
  const float* p   = (const float*)d_in[1];
  const float* s   = (const float*)d_in[2];
  const float* We1 = (const float*)d_in[3];
  const float* g1  = (const float*)d_in[4];
  const float* b1  = (const float*)d_in[5];
  const float* We2 = (const float*)d_in[6];
  const float* be2 = (const float*)d_in[7];
  const float* Wm  = (const float*)d_in[8];
  const float* bm  = (const float*)d_in[9];
  const float* Wx1 = (const float*)d_in[10];
  const float* bx1 = (const float*)d_in[11];
  const float* Wx2 = (const float*)d_in[12];
  const float* Wh1 = (const float*)d_in[13];
  const float* bh1 = (const float*)d_in[14];
  const float* gh  = (const float*)d_in[15];
  const float* bh  = (const float*)d_in[16];
  const float* Wh2 = (const float*)d_in[17];
  const float* bh2 = (const float*)d_in[18];

  float* ws = (float*)d_ws;
  float* ABrow = ws;                    // 4096*144   = 589824
  float* nd    = ws + 589824;           // 1M*2       = 2097152
  float* stats = ws + 2686976;          // 144
  float* a1c1  = ws + 2687120;          // 144
  float* maggW = ws + 2687264;          // 4096*72    = 294912
  float* yW    = ws + 2982176;          // 4096*72    = 294912
  float* a2c2  = ws + 3277088;          // 144

  float* hout = (float*)d_out;          // 16*256*72 = 294912
  float* pout = hout + 294912;          // 16*256*4  = 16384

  kA <<<64, 128, 0, stream>>>(h, We1, ABrow);
  kP <<<dim3(256, 16), 256, 0, stream>>>(p, nd, stats);
  kS <<<dim3(72, 64), 256, 0, stream>>>(ABrow, nd, We1, stats);
  kF1<<<1, 128, 0, stream>>>(stats, g1, b1, a1c1);
  kD <<<dim3(16, 16), 256, 0, stream>>>(p, ABrow, nd, We1, We2, be2, Wm, bm,
                                        Wx1, bx1, Wx2, a1c1, maggW, pout);
  kE1<<<64, 128, 0, stream>>>(h, maggW, s, Wh1, bh1, yW);
  kE2<<<72, 256, 0, stream>>>(yW, gh, bh, a2c2);
  kE4<<<64, 128, 0, stream>>>(h, yW, a2c2, Wh2, bh2, hout);
}
